// TABL_36721970381092
// MI455X (gfx1250) — matrix operand order repaired and checked by reading, not compiled, not run on hardware
//
#include <hip/hip_runtime.h>

typedef __attribute__((ext_vector_type(16))) __bf16 v16bf;
typedef __attribute__((ext_vector_type(8)))  float  v8f;

#define NB  256
#define ND1 512
#define NT  256
#define NO1 256
#define NO2 128

// Dynamic LDS layout:
//   [0      , 256 KB) : sE   float[NO1*NT]  (phase C/D reuses first 128KB as sY)
//   [256 KB , 272 KB) : sB buffer 0 (bf16 fragment staging, 16 KB)
//   [272 KB , 288 KB) : sB buffer 1 (bf16 fragment staging, 16 KB)
//   [288 KB , 289 KB) : red  float[256]
#define SMEM_BYTES (289 * 1024)
#define SB_ELEMS   8192   // bf16 elements per staging buffer (16 KB)

union BFrag { v16bf v; uint4 u[2]; };

// Stage a 32 x ncols f32 tile (row stride = gstride floats) from global into
// fragment-major bf16 LDS.  Slot s (16 B) holds, for (jt = s>>6, g = (s>>5)&1,
// lane = s&31): elements K = (lane>>4)*8 + g*16 + [0..7], N = jt*16 + (lane&15).
// A per-lane fragment read is then 2 contiguous ds_load_b128 (bank-conflict free).
__device__ __forceinline__ void stage_b_tile(__bf16* sB, const float* __restrict__ gsrc,
                                             int gstride, int ncols, int tid) {
    uint4* slots = (uint4*)sB;
    for (int s = tid; s < ncols * 4; s += 256) {
        int lane_w = s & 31;
        int g      = (s >> 5) & 1;
        int jt_w   = s >> 6;
        int n      = jt_w * 16 + (lane_w & 15);
        int kbase  = (lane_w >> 4) * 8 + g * 16;
        union { uint4 u; __bf16 h[8]; } pk;
        #pragma unroll
        for (int e = 0; e < 8; ++e)
            pk.h[e] = (__bf16)gsrc[(size_t)(kbase + e) * gstride + n];
        slots[s] = pk.u;
    }
}

__device__ __forceinline__ v16bf read_b_frag(const __bf16* sB, int jt, int lane) {
    BFrag f;
    const uint4* p = (const uint4*)sB;
    f.u[0] = p[(jt * 2 + 0) * 32 + lane];
    f.u[1] = p[(jt * 2 + 1) * 32 + lane];
    return f.v;
}

// A-fragment (16x32 bf16) from contiguous f32 row: lane&15 = row M; lanes 0-15
// hold K 0-7 / 16-23, lanes 16-31 hold K 8-15 / 24-31 (ISA 7.12.2).
__device__ __forceinline__ v16bf pack_a_f32(const float* __restrict__ src_row,
                                            int kbase, int lhalf) {
    v16bf a;
    const float* p = src_row + kbase + lhalf * 8;
    #pragma unroll
    for (int e = 0; e < 16; ++e) {
        int k = (e < 8) ? e : (e + 8);
        a[e] = (__bf16)p[k];
    }
    return a;
}

__global__ __launch_bounds__(256, 1)
void tabl_fused_kernel(const float* __restrict__ gx,  const float* __restrict__ gW1,
                       const float* __restrict__ gW,  const float* __restrict__ gW2,
                       const float* __restrict__ galpha, const float* __restrict__ gbias,
                       float* __restrict__ gout) {
    extern __shared__ char smem_raw[];
    float*  sE     = (float*)smem_raw;                       // 256x256 f32
    __bf16* sBbase = (__bf16*)(smem_raw + 256 * 1024);       // 2 x 16 KB ping-pong
    float*  red    = (float*)(smem_raw + 288 * 1024);

    const int b     = blockIdx.x;
    const int tid   = threadIdx.x;
    const int wave  = tid >> 5;      // 8 waves
    const int lane  = tid & 31;
    const int lhalf = lane >> 4;
    const int l16   = lane & 15;

    const float alpha = galpha[0];
    const float* xb = gx + (size_t)b * ND1 * NT;

    // ================= Phase A: E = W1 @ x[b]  (bf16 WMMA, f32 acc) ========
    for (int half = 0; half < 2; ++half) {
        v8f acc[2][8];
        #pragma unroll
        for (int rt = 0; rt < 2; ++rt)
            #pragma unroll
            for (int jt = 0; jt < 8; ++jt)
                #pragma unroll
                for (int r = 0; r < 8; ++r) acc[rt][jt][r] = 0.0f;

        stage_b_tile(sBbase, xb + half * 128, NT, 128, tid);   // kb = 0
        for (int kb = 0; kb < 16; ++kb) {
            __syncthreads();
            if (kb < 15)
                stage_b_tile(sBbase + ((kb + 1) & 1) * SB_ELEMS,
                             xb + (size_t)((kb + 1) * 32) * NT + half * 128,
                             NT, 128, tid);
            const __bf16* buf = sBbase + (kb & 1) * SB_ELEMS;

            v16bf afrag[2];
            #pragma unroll
            for (int rt = 0; rt < 2; ++rt) {
                int row = wave * 32 + rt * 16 + l16;
                afrag[rt] = pack_a_f32(gW1 + (size_t)row * ND1, kb * 32, lhalf);
            }
            // 1-deep software pipeline on B fragments: issue ds loads for jt+1
            // before the WMMAs of jt so LDS latency overlaps the matrix pipe.
            v16bf bcur = read_b_frag(buf, 0, lane);
            #pragma unroll
            for (int jt = 0; jt < 8; ++jt) {
                v16bf bnext;
                if (jt < 7) bnext = read_b_frag(buf, jt + 1, lane);
                #pragma unroll
                for (int rt = 0; rt < 2; ++rt)
                    acc[rt][jt] = __builtin_amdgcn_wmma_f32_16x16x32_bf16(
                        false, afrag[rt], false, bcur, (short)0, acc[rt][jt],
                        false, false);
                bcur = bnext;
            }
        }
        // E strip (f32) -> LDS.  C/D layout: VGPR r <-> row rt*16 + lhalf*8 + r
        #pragma unroll
        for (int rt = 0; rt < 2; ++rt)
            #pragma unroll
            for (int jt = 0; jt < 8; ++jt)
                #pragma unroll
                for (int r = 0; r < 8; ++r) {
                    int m = wave * 32 + rt * 16 + lhalf * 8 + r;
                    int n = half * 128 + jt * 16 + l16;
                    sE[m * NT + n] = acc[rt][jt][r];
                }
    }
    __syncthreads();

    // ======== Phase B: S = E @ W, row softmax, blended = E*(a+(1-a)A) ======
    {
        v8f s[2][16];
        #pragma unroll
        for (int rt = 0; rt < 2; ++rt)
            #pragma unroll
            for (int jt = 0; jt < 16; ++jt)
                #pragma unroll
                for (int r = 0; r < 8; ++r) s[rt][jt][r] = 0.0f;

        stage_b_tile(sBbase, gW, NT, NT, tid);                 // kt = 0
        for (int kt = 0; kt < 8; ++kt) {
            __syncthreads();
            if (kt < 7)
                stage_b_tile(sBbase + ((kt + 1) & 1) * SB_ELEMS,
                             gW + (size_t)((kt + 1) * 32) * NT, NT, NT, tid);
            const __bf16* buf = sBbase + (kt & 1) * SB_ELEMS;

            v16bf afrag[2];
            #pragma unroll
            for (int rt = 0; rt < 2; ++rt) {
                int row = wave * 32 + rt * 16 + l16;
                afrag[rt] = pack_a_f32(sE + row * NT, kt * 32, lhalf);
            }
            v16bf bcur = read_b_frag(buf, 0, lane);
            #pragma unroll
            for (int jt = 0; jt < 16; ++jt) {
                v16bf bnext;
                if (jt < 15) bnext = read_b_frag(buf, jt + 1, lane);
                #pragma unroll
                for (int rt = 0; rt < 2; ++rt)
                    s[rt][jt] = __builtin_amdgcn_wmma_f32_16x16x32_bf16(
                        false, afrag[rt], false, bcur, (short)0, s[rt][jt],
                        false, false);
                bcur = bnext;
            }
        }

        // softmax over N (256 cols) per row, then fold blend into sE
        #pragma unroll
        for (int rt = 0; rt < 2; ++rt)
            #pragma unroll
            for (int r = 0; r < 8; ++r) {
                int m = wave * 32 + rt * 16 + lhalf * 8 + r;
                float mx = -3.4e38f;
                #pragma unroll
                for (int jt = 0; jt < 16; ++jt) mx = fmaxf(mx, s[rt][jt][r]);
                #pragma unroll
                for (int off = 8; off >= 1; off >>= 1)
                    mx = fmaxf(mx, __shfl_xor(mx, off, 32));
                float sum = 0.0f;
                #pragma unroll
                for (int jt = 0; jt < 16; ++jt) sum += __expf(s[rt][jt][r] - mx);
                #pragma unroll
                for (int off = 8; off >= 1; off >>= 1)
                    sum += __shfl_xor(sum, off, 32);
                float inv = 1.0f / sum;
                #pragma unroll
                for (int jt = 0; jt < 16; ++jt) {
                    int n = jt * 16 + l16;
                    float a = __expf(s[rt][jt][r] - mx) * inv;
                    sE[m * NT + n] *= alpha + (1.0f - alpha) * a;
                }
            }
    }
    __syncthreads();

    // ================= Phase C: Y = blended @ W2 + bias ====================
    {
        v8f y[2][8];
        #pragma unroll
        for (int rt = 0; rt < 2; ++rt)
            #pragma unroll
            for (int jt = 0; jt < 8; ++jt)
                #pragma unroll
                for (int r = 0; r < 8; ++r) y[rt][jt][r] = 0.0f;

        stage_b_tile(sBbase, gW2, NO2, NO2, tid);              // kt = 0
        for (int kt = 0; kt < 8; ++kt) {
            __syncthreads();
            if (kt < 7)
                stage_b_tile(sBbase + ((kt + 1) & 1) * SB_ELEMS,
                             gW2 + (size_t)((kt + 1) * 32) * NO2, NO2, NO2, tid);
            const __bf16* buf = sBbase + (kt & 1) * SB_ELEMS;

            v16bf afrag[2];
            #pragma unroll
            for (int rt = 0; rt < 2; ++rt) {
                int row = wave * 32 + rt * 16 + l16;
                afrag[rt] = pack_a_f32(sE + row * NT, kt * 32, lhalf);
            }
            v16bf bcur = read_b_frag(buf, 0, lane);
            #pragma unroll
            for (int jt = 0; jt < 8; ++jt) {
                v16bf bnext;
                if (jt < 7) bnext = read_b_frag(buf, jt + 1, lane);
                #pragma unroll
                for (int rt = 0; rt < 2; ++rt)
                    y[rt][jt] = __builtin_amdgcn_wmma_f32_16x16x32_bf16(
                        false, afrag[rt], false, bcur, (short)0, y[rt][jt],
                        false, false);
                bcur = bnext;
            }
        }

        __syncthreads();          // everyone done reading sE before reuse as sY
        float* sY = sE;           // 256x128 f32 (128 KB)
        #pragma unroll
        for (int rt = 0; rt < 2; ++rt)
            #pragma unroll
            for (int jt = 0; jt < 8; ++jt)
                #pragma unroll
                for (int r = 0; r < 8; ++r) {
                    int m = wave * 32 + rt * 16 + lhalf * 8 + r;
                    int n = jt * 16 + l16;
                    sY[m * NO2 + n] = y[rt][jt][r] + gbias[m * NO2 + n];
                }
    }
    __syncthreads();

    // ======== Phase D: softmax over O1 (rows) per (b, o2) column ===========
    {
        const float* sY = sE;
        int c   = tid & 127;      // column
        int seg = tid >> 7;       // row segment (128 rows each)
        float mx = -3.4e38f;
        for (int r = 0; r < 128; ++r)
            mx = fmaxf(mx, sY[(seg * 128 + r) * NO2 + c]);
        red[seg * 128 + c] = mx;
        __syncthreads();
        float m2 = fmaxf(red[c], red[128 + c]);
        __syncthreads();
        float sum = 0.0f;
        for (int r = 0; r < 128; ++r)
            sum += __expf(sY[(seg * 128 + r) * NO2 + c] - m2);
        red[seg * 128 + c] = sum;
        __syncthreads();
        float inv = 1.0f / (red[c] + red[128 + c]);
        float* outb = gout + (size_t)b * NO1 * NO2;
        for (int r = 0; r < 128; ++r) {
            int m = seg * 128 + r;
            outb[m * NO2 + c] = __expf(sY[m * NO2 + c] - m2) * inv;
        }
    }
}

extern "C" void kernel_launch(void* const* d_in, const int* in_sizes, int n_in,
                              void* d_out, int out_size, void* d_ws, size_t ws_size,
                              hipStream_t stream) {
    (void)in_sizes; (void)n_in; (void)out_size; (void)d_ws; (void)ws_size;
    const float* x     = (const float*)d_in[0];
    const float* W1    = (const float*)d_in[1];
    const float* W     = (const float*)d_in[2];
    const float* W2    = (const float*)d_in[3];
    const float* alpha = (const float*)d_in[4];
    const float* bias  = (const float*)d_in[5];
    float* out = (float*)d_out;
    tabl_fused_kernel<<<NB, 256, SMEM_BYTES, stream>>>(x, W1, W, W2, alpha, bias, out);
}
